// MambaLayer_50895362458004
// MI455X (gfx1250) — matrix operand order repaired and checked by reading, not compiled, not run on hardware
//
#include <hip/hip_runtime.h>
#include <hip/hip_bf16.h>

typedef __bf16 bf16;
typedef __attribute__((ext_vector_type(16))) __bf16 v16bf;
typedef __attribute__((ext_vector_type(8)))  __bf16 v8bf;
typedef __attribute__((ext_vector_type(8)))  float  v8f;

#define L_SEQ 2048
#define C_DIM 384
#define D_IN  768
#define NCHUNK 32
#define CLEN   64   // L_SEQ / NCHUNK

__device__ __forceinline__ bf16 to_bf16(float f) {
    union { float f; unsigned u; } x; x.f = f;
    unsigned r = x.u + 0x7fffu + ((x.u >> 16) & 1u);   // round-to-nearest-even
    unsigned short h = (unsigned short)(r >> 16);
    return __builtin_bit_cast(bf16, h);
}

__device__ __forceinline__ float softplusf(float v) {
    return v > 20.f ? v : log1pf(__expf(v));
}

__device__ __forceinline__ float siluf(float v) {
    return v / (1.f + __expf(-v));
}

__device__ __forceinline__ v16bf cat16(v8bf lo, v8bf hi) {
    return __builtin_shufflevector(lo, hi, 0,1,2,3,4,5,6,7,8,9,10,11,12,13,14,15);
}

// direction index map: sequence position t -> source/dest position l
__device__ __forceinline__ int dir_idx(int dir, int t) {
    if (dir == 0) return t;                      // forward
    if (dir == 1) return (L_SEQ - 1) - t;        // backward (flip)
    return ((t & 7) << 8) + (t >> 3);            // slice: perm[t] = (t%8)*256 + t/8
}

// ---------------- LayerNorm over channels, emit bf16 (L x C) ----------------
__global__ __launch_bounds__(128)
void ln_kernel(const float* __restrict__ x, const float* __restrict__ g,
               const float* __restrict__ b, bf16* __restrict__ xn) {
    const int l = blockIdx.x;
    __shared__ float s1[128], s2[128];
    float v[3]; float sum = 0.f, sq = 0.f;
    #pragma unroll
    for (int i = 0; i < 3; ++i) {
        int c = threadIdx.x + i * 128;
        v[i] = x[(size_t)c * L_SEQ + l];
        sum += v[i]; sq += v[i] * v[i];
    }
    s1[threadIdx.x] = sum; s2[threadIdx.x] = sq; __syncthreads();
    for (int st = 64; st > 0; st >>= 1) {
        if (threadIdx.x < st) { s1[threadIdx.x] += s1[threadIdx.x + st];
                                s2[threadIdx.x] += s2[threadIdx.x + st]; }
        __syncthreads();
    }
    float mu  = s1[0] * (1.f / C_DIM);
    float var = s2[0] * (1.f / C_DIM) - mu * mu;
    float rs  = rsqrtf(var + 1e-5f);
    #pragma unroll
    for (int i = 0; i < 3; ++i) {
        int c = threadIdx.x + i * 128;
        xn[(size_t)l * C_DIM + c] = to_bf16((v[i] - mu) * rs * g[c] + b[c]);
    }
}

// ------------- f32 -> bf16 copy with zero padding (Rp x Cp from R x C) -------------
__global__ __launch_bounds__(256)
void cvt_pad_kernel(const float* __restrict__ src, bf16* __restrict__ dst,
                    int R, int C, int Rp, int Cp) {
    int i = blockIdx.x * 256 + threadIdx.x;
    if (i >= Rp * Cp) return;
    int r = i / Cp, c = i % Cp;
    dst[i] = to_bf16((r < R && c < C) ? src[r * C + c] : 0.f);
}

__global__ __launch_bounds__(256)
void cvt_kernel(const float* __restrict__ src, bf16* __restrict__ dst, int n) {
    int i = blockIdx.x * 256 + threadIdx.x;
    if (i < n) dst[i] = to_bf16(src[i]);
}

__global__ __launch_bounds__(256)
void combine_kernel(const float* __restrict__ y0, const float* __restrict__ y1,
                    const float* __restrict__ y2, bf16* __restrict__ yb, int n) {
    int i = blockIdx.x * 256 + threadIdx.x;
    if (i < n) yb[i] = to_bf16(y0[i] + y1[i] + y2[i]);
}

// ---------------- WMMA bf16 GEMM: C[m][n] = sum_k A[m][k] * Bw[n][k] ----------------
// Each wave computes a (16*MT) x (16*NT) tile: MT*NT independent accumulator chains,
// software-pipelined (double-buffered fragments): next k-step's 2*MT+2*NT b128 loads
// are issued BEFORE this step's WMMAs, so WMMAs wait only on loads from the previous
// iteration. At MT=2,NT=4: 12 loads feed 8 WMMAs (1.5 loads/WMMA).
// MODE 0: plain f32 store (ldc = N leading dim)
// MODE 1: softplus(v + bias[n]) store
// MODE 2: transposed store with residual: out[n*ldc + m] = v + resid[n*ldc + m]
template <int MODE, int MT, int NT>
__global__ __launch_bounds__(256)
void gemm_wmma(const bf16* __restrict__ A, int lda,
               const bf16* __restrict__ Bw, int ldb,
               float* __restrict__ C, int ldc,
               int M, int N, int K,
               const float* __restrict__ bias,
               const float* __restrict__ resid) {
    const int lane = threadIdx.x & 31;
    const int wave = threadIdx.x >> 5;
    const int m0 = blockIdx.y * (16 * MT);
    const int n0 = (blockIdx.x * 8 + wave) * (16 * NT);
    if (n0 >= N || m0 >= M) return;      // wave-uniform -> EXEC all ones for WMMA

    const int half = lane >> 4;          // 0 or 1
    const int idx  = lane & 15;          // A-row / B-col / D-col
    const bf16* Ap[MT];
    const bf16* Bp[NT];
    #pragma unroll
    for (int i = 0; i < MT; ++i)
        Ap[i] = A + (size_t)(m0 + i * 16 + idx) * lda + half * 8;    // K 0..7 / 8..15 (+16)
    #pragma unroll
    for (int j = 0; j < NT; ++j)
        Bp[j] = Bw + (size_t)(n0 + j * 16 + idx) * ldb + half * 8;   // K 0..7 / 8..15 (+16)

    v8f acc[MT][NT] = {};
    v8bf a0[MT], a1[MT], b0[NT], b1[NT];
    #pragma unroll
    for (int i = 0; i < MT; ++i) { a0[i] = *(const v8bf*)(Ap[i]);
                                   a1[i] = *(const v8bf*)(Ap[i] + 16); }
    #pragma unroll
    for (int j = 0; j < NT; ++j) { b0[j] = *(const v8bf*)(Bp[j]);
                                   b1[j] = *(const v8bf*)(Bp[j] + 16); }

    for (int k = 0; k < K; k += 32) {
        v16bf a[MT], b[NT];
        #pragma unroll
        for (int i = 0; i < MT; ++i) a[i] = cat16(a0[i], a1[i]);
        #pragma unroll
        for (int j = 0; j < NT; ++j) b[j] = cat16(b0[j], b1[j]);

        if (k + 32 < K) {   // prefetch next k-step before issuing this step's WMMAs
            #pragma unroll
            for (int i = 0; i < MT; ++i) { a0[i] = *(const v8bf*)(Ap[i] + k + 32);
                                           a1[i] = *(const v8bf*)(Ap[i] + k + 48); }
            #pragma unroll
            for (int j = 0; j < NT; ++j) { b0[j] = *(const v8bf*)(Bp[j] + k + 32);
                                           b1[j] = *(const v8bf*)(Bp[j] + k + 48); }
        }
        #pragma unroll
        for (int i = 0; i < MT; ++i)
            #pragma unroll
            for (int j = 0; j < NT; ++j)
                acc[i][j] = __builtin_amdgcn_wmma_f32_16x16x32_bf16(
                    /*neg_a=*/false, a[i], /*neg_b=*/false, b[j],
                    /*c_mod=*/(short)0, acc[i][j], /*reuse_a=*/false, /*reuse_b=*/false);
    }

    #pragma unroll
    for (int i = 0; i < MT; ++i) {
        #pragma unroll
        for (int j = 0; j < NT; ++j) {
            const int n = n0 + j * 16 + idx;
            #pragma unroll
            for (int e = 0; e < 8; ++e) {
                int m = m0 + i * 16 + half * 8 + e;
                float v = acc[i][j][e];
                if (MODE == 0) {
                    C[(size_t)m * ldc + n] = v;
                } else if (MODE == 1) {
                    C[(size_t)m * ldc + n] = softplusf(v + bias[n]);
                } else {
                    C[(size_t)n * ldc + m] = v + resid[(size_t)n * ldc + m];
                }
            }
        }
    }
}

// ------------- gather + depthwise conv(k=4) + SiLU, per direction -------------
__global__ __launch_bounds__(256)
void conv_silu_kernel(const float* __restrict__ xz,
                      const float* cwf, const float* cwb, const float* cws,
                      const float* cbf, const float* cbb, const float* cbs,
                      float* __restrict__ xc_all, bf16* __restrict__ xcb_all) {
    const int dir = blockIdx.y;
    const int gid = blockIdx.x * 256 + threadIdx.x;   // 0 .. L*D_IN-1
    const int t = gid / D_IN, d = gid % D_IN;
    const float* cw = dir == 0 ? cwf : (dir == 1 ? cwb : cws);
    const float* cb = dir == 0 ? cbf : (dir == 1 ? cbb : cbs);
    float acc = cb[d];
    #pragma unroll
    for (int j = 0; j < 4; ++j) {
        int u = t - 3 + j;
        if (u >= 0) {
            int l = dir_idx(dir, u);
            acc += cw[d * 4 + j] * xz[(size_t)l * (2 * D_IN) + d];
        }
    }
    float v = siluf(acc);
    size_t o = (size_t)dir * L_SEQ * D_IN + gid;
    xc_all[o]  = v;
    xcb_all[o] = to_bf16(v);
}

// --------------------- chunked (two-level) selective scan ---------------------
// h_t = a_t * h_{t-1} + dx_t * B_t is linear in h:
//   phase 1: per chunk (64 steps), compute P = prod(a) and local end-state (h0 = 0)
//   phase 2: serial combine over the 32 chunks -> true start state per chunk
//   phase 3: re-run chunks in parallel from true start states, emit gated outputs
// Critical path drops from 2048 steps to 64+32+64; 288 concurrent workgroups.

// state buffers layout: [dir][chunk][d][s]  (s fastest)
__global__ __launch_bounds__(256)
void scan_phase1(const float* __restrict__ dt_all, const float* __restrict__ xc_all,
                 const float* __restrict__ dbl_all,
                 const float* Af, const float* Ab, const float* As,
                 float* __restrict__ Pbuf, float* __restrict__ Hloc) {
    const int dir = blockIdx.z;
    const int c   = blockIdx.y;                       // chunk
    const int d   = blockIdx.x * 256 + threadIdx.x;   // channel
    const float* Alog = dir == 0 ? Af : (dir == 1 ? Ab : As);
    const float* dt   = dt_all  + (size_t)dir * L_SEQ * D_IN;
    const float* xc   = xc_all  + (size_t)dir * L_SEQ * D_IN;
    const float* dbl  = dbl_all + (size_t)dir * L_SEQ * 64;

    float Aneg[16], h[16], P[16];
    #pragma unroll
    for (int s = 0; s < 16; ++s) { Aneg[s] = -__expf(Alog[d * 16 + s]); h[s] = 0.f; P[s] = 1.f; }

    __shared__ float bs[CLEN][16];                    // B rows for this chunk
    #pragma unroll
    for (int j = 0; j < 4; ++j) {
        int lin = threadIdx.x + j * 256;
        int r = lin >> 4, cc = lin & 15;
        bs[r][cc] = dbl[(size_t)(c * CLEN + r) * 64 + 24 + cc];
    }
    __syncthreads();

    for (int u = 0; u < CLEN; ++u) {
        int t = c * CLEN + u;
        float dtv = dt[(size_t)t * D_IN + d];
        float dx  = dtv * xc[(size_t)t * D_IN + d];
        #pragma unroll
        for (int s = 0; s < 16; ++s) {
            float a = __expf(dtv * Aneg[s]);
            P[s] *= a;
            h[s] = a * h[s] + dx * bs[u][s];
        }
    }
    size_t o = (((size_t)dir * NCHUNK + c) * D_IN + d) * 16;
    #pragma unroll
    for (int s = 0; s < 16; ++s) { Pbuf[o + s] = P[s]; Hloc[o + s] = h[s]; }
}

__global__ __launch_bounds__(256)
void scan_phase2(const float* __restrict__ Pbuf, const float* __restrict__ Hloc,
                 float* __restrict__ Hstart) {
    int i = blockIdx.x * 256 + threadIdx.x;           // 0 .. 3*D_IN*16-1
    if (i >= 3 * D_IN * 16) return;
    int dir = i / (D_IN * 16);
    int ds  = i % (D_IN * 16);
    float h = 0.f;
    for (int c = 0; c < NCHUNK; ++c) {
        size_t o = ((size_t)dir * NCHUNK + c) * (D_IN * 16) + ds;
        Hstart[o] = h;
        h = Pbuf[o] * h + Hloc[o];
    }
}

__global__ __launch_bounds__(256)
void scan_phase3(const float* __restrict__ dt_all, const float* __restrict__ xc_all,
                 const float* __restrict__ dbl_all, const float* __restrict__ xz,
                 const float* Af, const float* Ab, const float* As,
                 const float* Df, const float* Db, const float* Ds,
                 const float* __restrict__ Hstart, float* __restrict__ ydir_all) {
    const int dir = blockIdx.z;
    const int c   = blockIdx.y;
    const int d   = blockIdx.x * 256 + threadIdx.x;
    const float* Alog = dir == 0 ? Af : (dir == 1 ? Ab : As);
    const float* Dp   = dir == 0 ? Df : (dir == 1 ? Db : Ds);
    const float* dt   = dt_all  + (size_t)dir * L_SEQ * D_IN;
    const float* xc   = xc_all  + (size_t)dir * L_SEQ * D_IN;
    const float* dbl  = dbl_all + (size_t)dir * L_SEQ * 64;
    float* ydir       = ydir_all + (size_t)dir * L_SEQ * D_IN;

    float Aneg[16], h[16];
    size_t o = (((size_t)dir * NCHUNK + c) * D_IN + d) * 16;
    #pragma unroll
    for (int s = 0; s < 16; ++s) { Aneg[s] = -__expf(Alog[d * 16 + s]); h[s] = Hstart[o + s]; }
    const float Dv = Dp[d];

    __shared__ float bc[CLEN][32];                    // B[0..15], C[16..31]
    #pragma unroll
    for (int j = 0; j < 8; ++j) {
        int lin = threadIdx.x + j * 256;
        int r = lin >> 5, cc = lin & 31;
        bc[r][cc] = dbl[(size_t)(c * CLEN + r) * 64 + 24 + cc];
    }
    __syncthreads();

    for (int u = 0; u < CLEN; ++u) {
        int t = c * CLEN + u;
        float dtv = dt[(size_t)t * D_IN + d];
        float xv  = xc[(size_t)t * D_IN + d];
        float dx  = dtv * xv;
        float y = 0.f;
        #pragma unroll
        for (int s = 0; s < 16; ++s) {
            float a = __expf(dtv * Aneg[s]);
            h[s] = a * h[s] + dx * bc[u][s];
            y += h[s] * bc[u][16 + s];
        }
        int l = dir_idx(dir, t);
        float zv = xz[(size_t)l * (2 * D_IN) + D_IN + d];
        ydir[(size_t)l * D_IN + d] = (y + xv * Dv) * siluf(zv);
    }
}

extern "C" void kernel_launch(void* const* d_in, const int* in_sizes, int n_in,
                              void* d_out, int out_size, void* d_ws, size_t ws_size,
                              hipStream_t stream) {
    const float* x        = (const float*)d_in[0];
    const float* ln_g     = (const float*)d_in[1];
    const float* ln_b     = (const float*)d_in[2];
    const float* in_projw = (const float*)d_in[3];
    const float* out_projw= (const float*)d_in[4];
    const float* conv_w[3], *conv_b[3], *xproj_w[3], *dt_w[3], *dt_b[3], *A_log[3], *Dvec[3];
    for (int s = 0; s < 3; ++s) {
        int o = 5 + 7 * s;
        conv_w[s]  = (const float*)d_in[o + 0];
        conv_b[s]  = (const float*)d_in[o + 1];
        xproj_w[s] = (const float*)d_in[o + 2];
        dt_w[s]    = (const float*)d_in[o + 3];
        dt_b[s]    = (const float*)d_in[o + 4];
        A_log[s]   = (const float*)d_in[o + 5];
        Dvec[s]    = (const float*)d_in[o + 6];
    }
    float* out = (float*)d_out;

    // ---- workspace layout ----
    char* base = (char*)d_ws;
    size_t off = 0;
    auto take = [&](size_t bytes) { size_t o = off; off += (bytes + 255) & ~(size_t)255; return o; };
    bf16*  xn_bf  = (bf16*)(base + take((size_t)L_SEQ * C_DIM * 2));
    bf16*  w_in   = (bf16*)(base + take((size_t)1536 * 384 * 2));
    bf16*  w_out  = (bf16*)(base + take((size_t)384 * 768 * 2));
    bf16*  w_xp   = (bf16*)(base + take((size_t)3 * 64 * 768 * 2));
    bf16*  w_dt   = (bf16*)(base + take((size_t)3 * 768 * 32 * 2));
    float* xz     = (float*)(base + take((size_t)L_SEQ * 1536 * 4));
    float* xc     = (float*)(base + take((size_t)3 * L_SEQ * D_IN * 4));
    bf16*  xc_bf  = (bf16*) (base + take((size_t)3 * L_SEQ * D_IN * 2));
    float* dbl    = (float*)(base + take((size_t)3 * L_SEQ * 64 * 4));
    bf16*  dbl_bf = (bf16*) (base + take((size_t)3 * L_SEQ * 64 * 2));
    float* dtbuf  = (float*)(base + take((size_t)3 * L_SEQ * D_IN * 4));
    float* ydir   = (float*)(base + take((size_t)3 * L_SEQ * D_IN * 4));
    bf16*  y_bf   = (bf16*) (base + take((size_t)L_SEQ * D_IN * 2));
    float* Pbuf   = (float*)(base + take((size_t)3 * NCHUNK * D_IN * 16 * 4));
    float* Hloc   = (float*)(base + take((size_t)3 * NCHUNK * D_IN * 16 * 4));
    float* Hstart = (float*)(base + take((size_t)3 * NCHUNK * D_IN * 16 * 4));

    // 1) LayerNorm -> bf16 (L x 384)
    ln_kernel<<<L_SEQ, 128, 0, stream>>>(x, ln_g, ln_b, xn_bf);

    // 2) weight conversions (bf16, zero-padded where needed)
    {
        int n;
        n = 1536 * 384;
        cvt_pad_kernel<<<(n + 255) / 256, 256, 0, stream>>>(in_projw, w_in, 1536, 384, 1536, 384);
        n = 384 * 768;
        cvt_pad_kernel<<<(n + 255) / 256, 256, 0, stream>>>(out_projw, w_out, 384, 768, 384, 768);
        for (int s = 0; s < 3; ++s) {
            n = 64 * 768;   // pad 56 -> 64 rows
            cvt_pad_kernel<<<(n + 255) / 256, 256, 0, stream>>>(xproj_w[s], w_xp + (size_t)s * 64 * 768,
                                                                56, 768, 64, 768);
            n = 768 * 32;   // pad K 24 -> 32 with zeros (makes dbl garbage cols harmless)
            cvt_pad_kernel<<<(n + 255) / 256, 256, 0, stream>>>(dt_w[s], w_dt + (size_t)s * 768 * 32,
                                                                768, 24, 768, 32);
        }
    }

    // 3) in_proj GEMM: (2048x384) x (1536x384)^T -> xz (2048x1536)
    gemm_wmma<0, 2, 4><<<dim3(1536 / 512, L_SEQ / 32), 256, 0, stream>>>(
        xn_bf, 384, w_in, 384, xz, 1536, L_SEQ, 1536, 384, nullptr, nullptr);

    // 4) gather + conv + silu per direction
    conv_silu_kernel<<<dim3((L_SEQ * D_IN) / 256, 3), 256, 0, stream>>>(
        xz, conv_w[0], conv_w[1], conv_w[2], conv_b[0], conv_b[1], conv_b[2], xc, xc_bf);

    // 5) xproj GEMMs: (2048x768) x (64x768)^T -> dbl (2048x64)   (NT=1 at N=64)
    for (int s = 0; s < 3; ++s)
        gemm_wmma<0, 2, 1><<<dim3(1, L_SEQ / 32), 256, 0, stream>>>(
            xc_bf + (size_t)s * L_SEQ * D_IN, 768,
            w_xp + (size_t)s * 64 * 768, 768,
            dbl + (size_t)s * L_SEQ * 64, 64,
            L_SEQ, 64, 768, nullptr, nullptr);

    cvt_kernel<<<(3 * L_SEQ * 64 + 255) / 256, 256, 0, stream>>>(dbl, dbl_bf, 3 * L_SEQ * 64);

    // 6) dt GEMMs with fused bias + softplus: (2048x32[lda 64]) x (768x32)^T -> dt (2048x768)
    for (int s = 0; s < 3; ++s)
        gemm_wmma<1, 2, 4><<<dim3(2, L_SEQ / 32), 256, 0, stream>>>(
            dbl_bf + (size_t)s * L_SEQ * 64, 64,
            w_dt + (size_t)s * 768 * 32, 32,
            dtbuf + (size_t)s * L_SEQ * D_IN, 768,
            L_SEQ, 768, 32, dt_b[s], nullptr);

    // 7) chunked selective scan: 3 phases
    scan_phase1<<<dim3(D_IN / 256, NCHUNK, 3), 256, 0, stream>>>(
        dtbuf, xc, dbl, A_log[0], A_log[1], A_log[2], Pbuf, Hloc);
    scan_phase2<<<(3 * D_IN * 16 + 255) / 256, 256, 0, stream>>>(Pbuf, Hloc, Hstart);
    scan_phase3<<<dim3(D_IN / 256, NCHUNK, 3), 256, 0, stream>>>(
        dtbuf, xc, dbl, xz,
        A_log[0], A_log[1], A_log[2], Dvec[0], Dvec[1], Dvec[2], Hstart, ydir);

    // 8) sum directions -> bf16; out_proj GEMM with fused transpose + residual
    combine_kernel<<<(L_SEQ * D_IN + 255) / 256, 256, 0, stream>>>(
        ydir, ydir + (size_t)L_SEQ * D_IN, ydir + (size_t)2 * L_SEQ * D_IN, y_bf, L_SEQ * D_IN);

    gemm_wmma<2, 2, 4><<<dim3(1, L_SEQ / 32), 256, 0, stream>>>(
        y_bf, 768, w_out, 768, out, L_SEQ, L_SEQ, 384, 768, nullptr, x);
}
